// TOGLWithSelfAttention_72413148611008
// MI455X (gfx1250) — compile-verified, matrix operand order checked by reading, not run on hardware
//
#include <hip/hip_runtime.h>
#include <hip/hip_bf16.h>
#include <math.h>

#define N_NODES 100000
#define N_EDGES 200000
#define G_GRAPHS 64
#define HID_D   256
#define BN_EPS  1e-5f

typedef __attribute__((ext_vector_type(16))) _Float16 v16h;
typedef __attribute__((ext_vector_type(8)))  float    v8f;

#define WMMA_F16(a, b, c) \
  __builtin_amdgcn_wmma_f32_16x16x32_f16(false, (a), false, (b), (short)0, (c), false, false)

// Fragment layout (ISA 7.12.2, 16-bit A 16x32 / B 32x16):
// lane<16: M(or N)=lane,  elems 0..7 = K k0..k0+7,   elems 8..15 = K k0+16..k0+23
// lane>=16: M(or N)=lane-16, elems 0..7 = K k0+8..15, elems 8..15 = K k0+24..31
// Packed buffers hold, per fragment, 16 contiguous f16 per lane (32B aligned):
//   packed[((chunkIdx)*32 + lane)*16 + i]

// ---------------- pack a row-major fp32 (K x C) weight matrix into B-fragment layout
__global__ void pack_b(const float* __restrict__ B, int K, int C, _Float16* __restrict__ out) {
  int nct = C >> 4, nkc = K >> 5;
  int total = nkc * nct * 32;
  for (int p = blockIdx.x * blockDim.x + threadIdx.x; p < total; p += gridDim.x * blockDim.x) {
    int lane = p & 31;
    int ct = (p >> 5) % nct;
    int kc = p / (32 * nct);
    int n = ct * 16 + (lane & 15);
    int kbase = kc * 32 + (lane >> 4) * 8;
    v16h f;
#pragma unroll
    for (int i = 0; i < 8; ++i) f[i] = (_Float16)B[(size_t)(kbase + i) * C + n];
#pragma unroll
    for (int i = 0; i < 8; ++i) f[8 + i] = (_Float16)B[(size_t)(kbase + 16 + i) * C + n];
    *(v16h*)(out + (size_t)p * 16) = f;
  }
}

// ---------------- pack staged fp32 tile (128 x ldk in LDS) into A-fragment f16 layout in LDS
static __device__ __forceinline__ void pack_a_tile(_Float16* dst, const float* stage, int ldk,
                                                   int nkc, int t) {
  int total = 8 * nkc * 32;  // 8 row-strips
  for (int p = t; p < total; p += 256) {
    int lane = p & 31;
    int kc = (p >> 5) % nkc;
    int s = p / (32 * nkc);
    int m = s * 16 + (lane & 15);
    int kbase = kc * 32 + (lane >> 4) * 8;
    const float* src = stage + m * ldk + kbase;
    v16h f;
#pragma unroll
    for (int i = 0; i < 8; ++i) f[i] = (_Float16)src[i];
#pragma unroll
    for (int i = 0; i < 8; ++i) f[8 + i] = (_Float16)src[16 + i];
    *(v16h*)(dst + (size_t)p * 16) = f;
  }
}

static __device__ __forceinline__ v16h frag_a(const _Float16* packA, int nkc, int wave, int kc, int lane) {
  return *(const v16h*)(packA + (size_t)((wave * nkc + kc) * 32 + lane) * 16);
}
static __device__ __forceinline__ v16h frag_b(const _Float16* packB, int nct, int kc, int ct, int lane) {
  return *(const v16h*)(packB + (size_t)((kc * nct + ct) * 32 + lane) * 16);
}

// ---------------- attention algebra precompute: M = W A W^T, u = W A b, v = b A W^T, c = b A b
__global__ void att_precompute(const float* __restrict__ W, const float* __restrict__ A,
                               const float* __restrict__ b,
                               float* M, float* u, float* v, float* cc) {
  extern __shared__ float sm[];  // T: 128*128, vtmp: 128
  float* T = sm;
  float* vtmp = sm + 128 * 128;
  int h = blockIdx.x;
  const float* Wh = W + h * 128 * 128;
  const float* Ah = A + h * 128 * 128;
  const float* bh = b + h * 128;
  int t = threadIdx.x;
  for (int idx = t; idx < 128 * 128; idx += blockDim.x) {
    int i = idx >> 7, j = idx & 127;
    float s = 0.f;
    for (int k = 0; k < 128; ++k) s += Wh[i * 128 + k] * Ah[k * 128 + j];
    T[idx] = s;
  }
  if (t < 128) {
    float s = 0.f;
    for (int k = 0; k < 128; ++k) s += bh[k] * Ah[k * 128 + t];
    vtmp[t] = s;
  }
  __syncthreads();
  for (int idx = t; idx < 128 * 128; idx += blockDim.x) {
    int i = idx >> 7, j = idx & 127;
    float s = 0.f;
    for (int k = 0; k < 128; ++k) s += T[i * 128 + k] * Wh[j * 128 + k];
    M[h * 128 * 128 + idx] = s;
  }
  if (t < 128) {
    float su = 0.f, sv = 0.f;
    for (int k = 0; k < 128; ++k) { su += T[t * 128 + k] * bh[k]; sv += vtmp[k] * Wh[t * 128 + k]; }
    u[h * 128 + t] = su;
    v[h * 128 + t] = sv;
  }
  if (t == 0) {
    float s = 0.f;
    for (int k = 0; k < 128; ++k) s += vtmp[k] * bh[k];
    cc[h] = s;
  }
}

// ---------------- q_h = x @ M_h for all 4 heads; one 128-row tile per block, staged x reused 4x
__global__ void gemm_q(const float* __restrict__ x, const _Float16* __restrict__ packM,
                       float* __restrict__ q, int n_nodes) {
  extern __shared__ float smf[];
  _Float16* packA = (_Float16*)smf;   // [0, 32KB): 8 strips * 4 kc * 32 lanes * 16 f16
  float* stage = smf + 8192;          // [32KB, 96KB): 128x128 fp32
  int row0 = blockIdx.x * 128;
  int t = threadIdx.x;
  for (int idx = t; idx < 128 * 128; idx += 256) {
    int r = idx >> 7, c = idx & 127;
    int gr = row0 + r;
    stage[idx] = (gr < n_nodes) ? x[(size_t)gr * 128 + c] : 0.f;
  }
  __syncthreads();
  pack_a_tile(packA, stage, 128, 4, t);
  __syncthreads();
  int wave = t >> 5, lane = t & 31;
  int mb = wave * 16 + ((lane >= 16) ? 8 : 0);
  for (int h = 0; h < 4; ++h) {
    const _Float16* Bp = packM + (size_t)h * 16384;  // 1024 frags * 16
    v8f zero = {};
    v8f acc[8];
#pragma unroll
    for (int i = 0; i < 8; ++i) acc[i] = zero;
#pragma unroll
    for (int kc = 0; kc < 4; ++kc) {
      v16h a = frag_a(packA, 4, wave, kc, lane);
#pragma unroll
      for (int ct = 0; ct < 8; ++ct) {
        v16h bf = frag_b(Bp, 8, kc, ct, lane);
        acc[ct] = WMMA_F16(a, bf, acc[ct]);
      }
    }
    float* qh = q + (size_t)h * n_nodes * 128;
#pragma unroll
    for (int ct = 0; ct < 8; ++ct) {
      int n = ct * 16 + (lane & 15);
#pragma unroll
      for (int j = 0; j < 8; ++j) {
        int gm = row0 + mb + j;
        if (gm < n_nodes) qh[(size_t)gm * 128 + n] = acc[ct][j];
      }
    }
  }
}

// ---------------- per-node scalars: asum[n] = sum_h x[n].u_h, bsum[n] = sum_h x[n].v_h
__global__ void node_scalars(const float* __restrict__ x, const float* __restrict__ u,
                             const float* __restrict__ v, float* asum, float* bsum, int n_nodes) {
  int gid = blockIdx.x * 8 + (threadIdx.x >> 5);
  int lane = threadIdx.x & 31;
  if (gid >= n_nodes) return;
  float4 x4 = ((const float4*)(x + (size_t)gid * 128))[lane];
  float a = 0.f, b = 0.f;
#pragma unroll
  for (int h = 0; h < 4; ++h) {
    float4 u4 = ((const float4*)(u + h * 128))[lane];
    float4 v4 = ((const float4*)(v + h * 128))[lane];
    a += x4.x * u4.x + x4.y * u4.y + x4.z * u4.z + x4.w * u4.w;
    b += x4.x * v4.x + x4.y * v4.y + x4.z * v4.z + x4.w * v4.w;
  }
  for (int off = 16; off; off >>= 1) { a += __shfl_xor(a, off, 32); b += __shfl_xor(b, off, 32); }
  if (lane == 0) { asum[gid] = a; bsum[gid] = b; }
}

// ---------------- edge scores: att[e] = sigmoid(mean_h(q_h[row].x[col] + a_h[row] + b_h[col] + c_h))
__global__ void edge_scores(const float* __restrict__ x, const float* __restrict__ q,
                            const int* __restrict__ ei, const float* __restrict__ asum,
                            const float* __restrict__ bsum, const float* __restrict__ cc,
                            float* __restrict__ att, int n_edges, int n_nodes) {
  int e = blockIdx.x * 8 + (threadIdx.x >> 5);
  int lane = threadIdx.x & 31;
  if (e >= n_edges) return;
  int r = ei[e];
  int c = ei[n_edges + e];
  float4 xc = ((const float4*)(x + (size_t)c * 128))[lane];
  float acc = 0.f;
#pragma unroll
  for (int h = 0; h < 4; ++h) {
    float4 qr = ((const float4*)(q + ((size_t)h * n_nodes + r) * 128))[lane];
    acc += qr.x * xc.x + qr.y * xc.y + qr.z * xc.z + qr.w * xc.w;
  }
  for (int off = 16; off; off >>= 1) acc += __shfl_xor(acc, off, 32);
  if (lane == 0) {
    float tot = (acc + asum[r] + bsum[c] + cc[0] + cc[1] + cc[2] + cc[3]) * 0.25f;
    att[e] = 1.f / (1.f + expf(-tot));
  }
}

// ---------------- fused: h = relu(x@Wf1+bf1); fv = h@Wf2+bf2; x0a = relu(fv@fold(Wl0)+bl0)
__global__ void filtration_fused(const float* __restrict__ x, const _Float16* __restrict__ packWf1,
                                 const float* __restrict__ bf1, const float* __restrict__ Wf2,
                                 const float* __restrict__ bf2, const float* __restrict__ Wl0,
                                 const float* __restrict__ bl0, float* __restrict__ x0a,
                                 int n_nodes) {
  extern __shared__ float smf[];
  // phase1: packA f16 [0,32KB), stage fp32 [32KB,96KB)
  // phase2 (after WMMA): hidden fp32 [0,128KB), fv [128KB,132KB)
  _Float16* packA = (_Float16*)smf;
  float* stage = smf + 8192;
  float* fv = smf + 32768;
  int row0 = blockIdx.x * 128;
  int t = threadIdx.x;
  for (int idx = t; idx < 128 * 128; idx += 256) {
    int r = idx >> 7, c = idx & 127;
    int gr = row0 + r;
    stage[idx] = (gr < n_nodes) ? x[(size_t)gr * 128 + c] : 0.f;
  }
  __syncthreads();
  pack_a_tile(packA, stage, 128, 4, t);
  __syncthreads();
  int wave = t >> 5, lane = t & 31;
  v8f zero = {};
  v8f acc[16];
#pragma unroll
  for (int i = 0; i < 16; ++i) acc[i] = zero;
#pragma unroll
  for (int kc = 0; kc < 4; ++kc) {
    v16h a = frag_a(packA, 4, wave, kc, lane);
#pragma unroll
    for (int ct = 0; ct < 16; ++ct) {
      v16h bf = frag_b(packWf1, 16, kc, ct, lane);
      acc[ct] = WMMA_F16(a, bf, acc[ct]);
    }
  }
  __syncthreads();  // all waves done with packA/stage; reuse LDS for hidden
  int mb = wave * 16 + ((lane >= 16) ? 8 : 0);
#pragma unroll
  for (int ct = 0; ct < 16; ++ct) {
    int n = ct * 16 + (lane & 15);
    float bias = bf1[n];
#pragma unroll
    for (int j = 0; j < 8; ++j) {
      float vv = acc[ct][j] + bias;
      smf[(mb + j) * HID_D + n] = vv > 0.f ? vv : 0.f;
    }
  }
  __syncthreads();
  // fv = h @ Wf2 + bf2 : each thread does 4 of the 8 filtration cols for one row
  {
    int m = t & 127;
    int jg = t >> 7;
    float s0 = 0.f, s1 = 0.f, s2 = 0.f, s3 = 0.f;
    for (int k = 0; k < HID_D; ++k) {
      float hv = smf[m * HID_D + k];
      const float* w = Wf2 + k * 8 + jg * 4;
      s0 += hv * w[0]; s1 += hv * w[1]; s2 += hv * w[2]; s3 += hv * w[3];
    }
    fv[m * 8 + jg * 4 + 0] = s0 + bf2[jg * 4 + 0];
    fv[m * 8 + jg * 4 + 1] = s1 + bf2[jg * 4 + 1];
    fv[m * 8 + jg * 4 + 2] = s2 + bf2[jg * 4 + 2];
    fv[m * 8 + jg * 4 + 3] = s3 + bf2[jg * 4 + 3];
  }
  __syncthreads();
  // x0a = relu(repeat(fv,2) @ Wl0 + bl0) == relu(fv @ (Wl0[2k]+Wl0[2k+1]) + bl0)
  {
    int m = t >> 1;
    int c0 = (t & 1) * 32;
    int gm = row0 + m;
    for (int c = c0; c < c0 + 32; ++c) {
      float s = 0.f;
#pragma unroll
      for (int k = 0; k < 8; ++k)
        s += fv[m * 8 + k] * (Wl0[(2 * k) * 64 + c] + Wl0[(2 * k + 1) * 64 + c]);
      s += bl0[c];
      s = s > 0.f ? s : 0.f;
      if (gm < n_nodes) x0a[(size_t)gm * 64 + c] = s;
    }
  }
}

// ---------------- segment sums (+counts) via atomics
__global__ void seg_reduce(const float* __restrict__ V, const int* __restrict__ batch,
                           float* S, float* cnt, int n, int C) {
  int idx = blockIdx.x * blockDim.x + threadIdx.x;
  int stride = gridDim.x * blockDim.x;
  int total = n * C;
  for (; idx < total; idx += stride) {
    int nd = idx / C, c = idx - nd * C;
    int g = batch[nd];
    atomicAdd(&S[g * C + c], V[idx]);
    if (c == 0) atomicAdd(&cnt[g], 1.f);
  }
}

// ---------------- XM = (S / max(cnt,1)) @ L  (G x K @ K x C)
__global__ void mean_gemm(const float* __restrict__ S, const float* __restrict__ cnt,
                          const float* __restrict__ L, float* XM, int K, int C) {
  int idx = blockIdx.x * blockDim.x + threadIdx.x;
  if (idx >= G_GRAPHS * C) return;
  int g = idx / C, c = idx - g * C;
  float ic = 1.f / fmaxf(cnt[g], 1.f);
  float s = 0.f;
  for (int k = 0; k < K; ++k) s += S[g * K + k] * ic * L[k * C + c];
  XM[idx] = s;
}

// ---------------- Vout = relu(Vin @ W + bias - xm[batch])  (K=64, C in {64,128}), WMMA
__global__ void setfn_gemm(const float* __restrict__ Vin, const _Float16* __restrict__ packW,
                           const float* __restrict__ bias, const float* __restrict__ xm,
                           const int* __restrict__ batch, float* __restrict__ Vout,
                           int n_nodes, int C) {
  extern __shared__ float smf[];
  _Float16* packA = (_Float16*)smf;        // [0,16KB): 8 strips * 2 kc * 32 * 16 f16
  float* stage = smf + 4096;               // [16KB,48KB): 128x64 fp32
  int* bloc = (int*)(smf + 12288);         // [48KB,+512B)
  int row0 = blockIdx.x * 128;
  int t = threadIdx.x;
  for (int idx = t; idx < 128 * 64; idx += 256) {
    int r = idx >> 6, c = idx & 63;
    int gr = row0 + r;
    stage[idx] = (gr < n_nodes) ? Vin[(size_t)gr * 64 + c] : 0.f;
  }
  if (t < 128) { int gr = row0 + t; bloc[t] = (gr < n_nodes) ? batch[gr] : 0; }
  __syncthreads();
  pack_a_tile(packA, stage, 64, 2, t);
  __syncthreads();
  int wave = t >> 5, lane = t & 31;
  int nct = C >> 4;  // 4 or 8 column tiles
  v8f zero = {};
  v8f acc[8];
#pragma unroll
  for (int i = 0; i < 8; ++i) acc[i] = zero;
#pragma unroll
  for (int kc = 0; kc < 2; ++kc) {
    v16h a = frag_a(packA, 2, wave, kc, lane);
    for (int ct = 0; ct < nct; ++ct) {
      v16h bf = frag_b(packW, nct, kc, ct, lane);
      acc[ct] = WMMA_F16(a, bf, acc[ct]);
    }
  }
  int mb = wave * 16 + ((lane >= 16) ? 8 : 0);
  for (int ct = 0; ct < nct; ++ct) {
    int n = ct * 16 + (lane & 15);
    float bn = bias[n];
#pragma unroll
    for (int j = 0; j < 8; ++j) {
      int ml = mb + j;
      int gm = row0 + ml;
      float vv = acc[ct][j] + bn - xm[bloc[ml] * C + n];
      vv = vv > 0.f ? vv : 0.f;
      if (gm < n_nodes) Vout[(size_t)gm * C + n] = vv;
    }
  }
}

// ---------------- BN stats: per-block column partials, 2 atomics per block per col
__global__ void bn_stats(const float* __restrict__ hid, float* s, float* sq, int n) {
  int c = threadIdx.x;  // 128 threads == 128 cols
  float a = 0.f, b = 0.f;
  for (int r = blockIdx.x; r < n; r += gridDim.x) {
    float v = hid[(size_t)r * 128 + c];
    a += v;
    b += v * v;
  }
  atomicAdd(&s[c], a);
  atomicAdd(&sq[c], b);
}

__global__ void bn_scale(const float* s, const float* sq, const float* gamma, const float* beta,
                         float* scale, float* shift, int n) {
  int c = threadIdx.x;
  float inv_n = 1.f / (float)n;
  float mu = s[c] * inv_n;
  float var = sq[c] * inv_n - mu * mu;
  float sc = gamma[c] * rsqrtf(var + BN_EPS);
  scale[c] = sc;
  shift[c] = beta[c] - mu * sc;
}

__global__ void bn_final(const float* __restrict__ x, const float* __restrict__ hid,
                         const float* __restrict__ scale, const float* __restrict__ shift,
                         float* __restrict__ out, int n) {
  int idx = blockIdx.x * blockDim.x + threadIdx.x;
  if (idx >= n * 128) return;
  int c = idx & 127;
  out[idx] = x[idx] + hid[idx] * scale[c] + shift[c];
}

extern "C" void kernel_launch(void* const* d_in, const int* in_sizes, int n_in,
                              void* d_out, int out_size, void* d_ws, size_t ws_size,
                              hipStream_t stream) {
  const int N = N_NODES, E = N_EDGES;
  const float* x   = (const float*)d_in[0];
  const int* ei    = (const int*)d_in[1];
  const int* batch = (const int*)d_in[2];
  const float* Wf1 = (const float*)d_in[3];
  const float* bf1 = (const float*)d_in[4];
  const float* Wf2 = (const float*)d_in[5];
  const float* bf2 = (const float*)d_in[6];
  const float* Wl0 = (const float*)d_in[7];
  const float* bl0 = (const float*)d_in[8];
  const float* G1  = (const float*)d_in[9];
  const float* bg1 = (const float*)d_in[10];
  const float* L1  = (const float*)d_in[11];
  const float* G2  = (const float*)d_in[12];
  const float* bg2 = (const float*)d_in[13];
  const float* L2  = (const float*)d_in[14];
  const float* gamma = (const float*)d_in[15];
  const float* beta  = (const float*)d_in[16];
  const float* Watt  = (const float*)d_in[17];
  const float* batt  = (const float*)d_in[18];
  const float* Aatt  = (const float*)d_in[19];

  float* out = (float*)d_out;
  float* att = out + (size_t)N * 128;

  // workspace layout (float units, regions 128B-aligned)
  float* ws = (float*)d_ws;
  size_t o = 0;
  float* M    = ws + o; o += (size_t)4 * 128 * 128;
  float* u    = ws + o; o += 4 * 128;
  float* v    = ws + o; o += 4 * 128;
  float* cc   = ws + o; o += 64;
  float* q    = ws + o; o += (size_t)4 * N * 128;
  float* asum = ws + o; o += N;
  float* bsum = ws + o; o += N;
  float* x0a  = ws + o; o += (size_t)N * 64;
  float* x0b  = ws + o; o += (size_t)N * 64;
  float* hid  = ws + o; o += (size_t)N * 128;
  float* stats = ws + o;  // S1(4096) cnt1(64) S2(4096) cnt2(64) bnS(128) bnSQ(128)
  float* S1 = stats;
  float* cnt1 = stats + 4096;
  float* S2 = stats + 4160;
  float* cnt2 = stats + 8256;
  float* bnS = stats + 8320;
  float* bnSQ = stats + 8448;
  o += 8576;
  float* xm1 = ws + o; o += 64 * 64;
  float* xm2 = ws + o; o += 64 * 128;
  float* bnScale = ws + o; o += 128;
  float* bnShift = ws + o; o += 128;
  _Float16* packM   = (_Float16*)(ws + o); o += 32768;  // 4 heads * 16384 f16
  _Float16* packWf1 = (_Float16*)(ws + o); o += 16384;  // 32768 f16
  _Float16* packG1  = (_Float16*)(ws + o); o += 2048;   // 4096 f16
  _Float16* packG2  = (_Float16*)(ws + o); o += 4096;   // 8192 f16

  hipMemsetAsync(stats, 0, 8576 * sizeof(float), stream);

  int tiles = (N + 127) / 128;

  // ---- pack weights into f16 WMMA fragment layout (once per call; L2-resident)
  pack_b<<<8, 256, 0, stream>>>(Wf1, 128, 256, packWf1);
  pack_b<<<1, 256, 0, stream>>>(G1, 64, 64, packG1);
  pack_b<<<2, 256, 0, stream>>>(G2, 64, 128, packG2);

  // ---- attention branch
  att_precompute<<<4, 256, (128 * 128 + 128) * sizeof(float), stream>>>(Watt, Aatt, batt, M, u, v, cc);
  for (int h = 0; h < 4; ++h)
    pack_b<<<4, 256, 0, stream>>>(M + h * 128 * 128, 128, 128, packM + (size_t)h * 16384);
  gemm_q<<<tiles, 256, 96 * 1024, stream>>>(x, packM, q, N);
  node_scalars<<<(N + 7) / 8, 256, 0, stream>>>(x, u, v, asum, bsum, N);
  edge_scores<<<(E + 7) / 8, 256, 0, stream>>>(x, q, ei, asum, bsum, cc, att, E, N);

  // ---- filtration + set_fn branch
  filtration_fused<<<tiles, 256, 132 * 1024 + 4096, stream>>>(
      x, packWf1, bf1, Wf2, bf2, Wl0, bl0, x0a, N);
  seg_reduce<<<1024, 256, 0, stream>>>(x0a, batch, S1, cnt1, N, 64);
  mean_gemm<<<(64 * 64 + 255) / 256, 256, 0, stream>>>(S1, cnt1, L1, xm1, 64, 64);
  setfn_gemm<<<tiles, 256, 48 * 1024 + 512, stream>>>(x0a, packG1, bg1, xm1, batch, x0b, N, 64);
  seg_reduce<<<1024, 256, 0, stream>>>(x0b, batch, S2, cnt2, N, 64);
  mean_gemm<<<(64 * 128 + 255) / 256, 256, 0, stream>>>(S2, cnt2, L2, xm2, 64, 128);
  setfn_gemm<<<tiles, 256, 48 * 1024 + 512, stream>>>(x0b, packG2, bg2, xm2, batch, hid, N, 128);

  // ---- batchnorm + residual
  bn_stats<<<256, 128, 0, stream>>>(hid, bnS, bnSQ, N);
  bn_scale<<<1, 128, 0, stream>>>(bnS, bnSQ, gamma, beta, bnScale, bnShift, N);
  bn_final<<<(N * 128 + 255) / 256, 256, 0, stream>>>(x, hid, bnScale, bnShift, out, N);
}
